// WithAttention_87402584473990
// MI455X (gfx1250) — matrix operand order repaired and checked by reading, not compiled, not run on hardware
//
#include <hip/hip_runtime.h>
#include <cstdint>

// ---------------------------------------------------------------------------
// Shapes (compile-time constants from the reference)
// ---------------------------------------------------------------------------
#define BB   64
#define TT   1024
#define EE   512
#define HH   512
#define AA   512
#define DIN  512
#define KX   1024   // DIN + EE
#define G3   1536   // 3*H

typedef __attribute__((ext_vector_type(16))) __bf16 bf16x16;
typedef __attribute__((ext_vector_type(8)))  float  v8f;
typedef __attribute__((ext_vector_type(4)))  uint32_t u32x4;
typedef __attribute__((ext_vector_type(8)))  uint32_t u32x8;

union BF16Frag { bf16x16 v; unsigned short s[16]; };

#if __has_builtin(__builtin_amdgcn_tanhf)
#define FAST_TANH(x) __builtin_amdgcn_tanhf(x)
#else
#define FAST_TANH(x) tanhf(x)
#endif

// fp32 -> bf16 round-to-nearest-even
static __device__ __forceinline__ unsigned short f2bf(float f) {
    unsigned u = __float_as_uint(f);
    u = u + 0x7FFFu + ((u >> 16) & 1u);
    return (unsigned short)(u >> 16);
}

// Index (in bf16 elements) of element (m, k) of a 16-row A-strip stored in
// contiguous-k A-fragment order (this code's choice; B loads match):
//   lane l = m + 16*half holds K[16*half .. 16*half+15] (v0..v7)
// Per k-step s (32 K), layout is [s][lane][chunk][j] => lane-contiguous 32B.
static __device__ __forceinline__ int afrag_index(int m, int k) {
    int s  = k >> 5;
    int kk = k & 31;
    int c  = kk >> 4;        // lane half
    int r8 = kk & 15;
    int hf = r8 >> 3;        // 8-elem chunk within lane
    int j  = r8 & 7;
    return s * 512 + (m + (c << 4)) * 16 + r8;
}

// ---------------------------------------------------------------------------
// 1) Wt[a][e] = bf16(W_enc[e][a])  (columns of W_enc become contiguous rows:
//    exactly what the 16-bit WMMA B-fragment wants)
// ---------------------------------------------------------------------------
__global__ __launch_bounds__(256) void wa_prep_wt(const float* __restrict__ W_enc,
                                                  unsigned short* __restrict__ Wt) {
    int gid = blockIdx.x * 256 + threadIdx.x;      // 262144 total
    int a = gid >> 9, e = gid & 511;
    Wt[(size_t)a * EE + e] = f2bf(W_enc[(size_t)e * AA + a]);
}

// ---------------------------------------------------------------------------
// 2) st_proj[b][a] = sum_h states[b][h] * W_state[h][a]   (tiny)
// ---------------------------------------------------------------------------
__global__ __launch_bounds__(256) void wa_stproj(const float* __restrict__ states,
                                                 const float* __restrict__ W_state,
                                                 float* __restrict__ stp) {
    int gid = blockIdx.x * 256 + threadIdx.x;      // 32768 total
    int b = gid >> 9, a = gid & 511;
    float acc = 0.f;
    #pragma unroll 4
    for (int h = 0; h < HH; ++h)
        acc = fmaf(states[b * HH + h], W_state[(size_t)h * AA + a], acc);
    stp[gid] = acc;
}

// ---------------------------------------------------------------------------
// 3) Fused scores kernel: scores[b,t] = sum_a v[a]*tanh((enc@W_enc)[b,t,a]+st[b,a])
//    Block = 128 thr = 4 waves, 64 rows of encoded staged to LDS as bf16
//    A-fragments; 32 n-tiles x 16 k-steps of v_wmma_f32_16x16x32_bf16.
// ---------------------------------------------------------------------------
__global__ __launch_bounds__(128) void wa_scores(const float* __restrict__ encoded,
                                                 const unsigned short* __restrict__ Wt,
                                                 const float* __restrict__ stp,
                                                 const float* __restrict__ vvec,
                                                 float* __restrict__ scores) {
    __shared__ alignas(32) unsigned short Als[4 * 8192];  // 4 strips x 16 rows x 512 K (64KB)

    const int tid        = threadIdx.x;
    const int block_base = blockIdx.x * 64;               // first global row
    const int b          = block_base >> 10;              // 64 | 1024 -> constant per block

    // ---- stage: encoded f32 -> bf16 in fragment order (read 64x512 once) ----
    const float2* src = (const float2*)(encoded + (size_t)block_base * EE);
    #pragma unroll 4
    for (int it = 0; it < 128; ++it) {
        int pidx = it * 128 + tid;                        // 16384 f32-pairs
        int row  = pidx >> 8;
        int k    = (pidx & 255) * 2;
        float2 f = src[(size_t)row * 256 + (k >> 1)];
        uint32_t pk = (uint32_t)f2bf(f.x) | ((uint32_t)f2bf(f.y) << 16);
        int li = (row >> 4) * 8192 + afrag_index(row & 15, k);  // li even
        *(uint32_t*)&Als[li] = pk;
    }
    __syncthreads();

    const int w = tid >> 5, lane = tid & 31;
    const int hf = lane >> 4, nl = lane & 15;
    const unsigned short* strip = &Als[w * 8192];

    float p[8];
    #pragma unroll
    for (int r = 0; r < 8; ++r) p[r] = 0.f;

    for (int nt = 0; nt < 32; ++nt) {
        const int n   = nt * 16 + nl;                     // column handled by this lane
        const float stv = stp[b * AA + n];
        const float vv  = vvec[n];
        v8f c = {};
        const unsigned short* wrow = Wt + (size_t)n * EE + hf * 16;
        #pragma unroll
        for (int s = 0; s < 16; ++s) {
            const bf16x16 a  = *(const bf16x16*)&strip[s * 512 + lane * 16];
            const bf16x16 bm = *(const bf16x16*)&wrow[s * 32];
            c = __builtin_amdgcn_wmma_f32_16x16x32_bf16(false, a, false, bm,
                                                        (short)0, c, false, false);
        }
        #pragma unroll
        for (int r = 0; r < 8; ++r)
            p[r] += vv * FAST_TANH(c[r] + stv);
    }

    // reduce the 16 lanes of each half (row m = r + 8*hf lives in one half)
    #pragma unroll
    for (int r = 0; r < 8; ++r) {
        float t = p[r];
        t += __shfl_xor(t, 1, 32);
        t += __shfl_xor(t, 2, 32);
        t += __shfl_xor(t, 4, 32);
        t += __shfl_xor(t, 8, 32);
        p[r] = t;
    }
    if (nl == 0) {
        int t0 = (block_base & (TT - 1)) + w * 16 + hf * 8;
        #pragma unroll
        for (int r = 0; r < 8; ++r)
            scores[(size_t)b * TT + t0 + r] = p[r];
    }
}

// ---------------------------------------------------------------------------
// 4) softmax over T=1024 per batch row
// ---------------------------------------------------------------------------
__global__ __launch_bounds__(256) void wa_softmax(const float* __restrict__ scores,
                                                  float* __restrict__ weights) {
    __shared__ float red[256];
    const int b = blockIdx.x, tid = threadIdx.x;
    const float* s = scores + (size_t)b * TT;
    float v0 = s[tid], v1 = s[tid + 256], v2 = s[tid + 512], v3 = s[tid + 768];
    float m = fmaxf(fmaxf(v0, v1), fmaxf(v2, v3));
    red[tid] = m; __syncthreads();
    for (int o = 128; o > 0; o >>= 1) { if (tid < o) red[tid] = fmaxf(red[tid], red[tid + o]); __syncthreads(); }
    m = red[0]; __syncthreads();
    float e0 = __expf(v0 - m), e1 = __expf(v1 - m), e2 = __expf(v2 - m), e3 = __expf(v3 - m);
    red[tid] = e0 + e1 + e2 + e3; __syncthreads();
    for (int o = 128; o > 0; o >>= 1) { if (tid < o) red[tid] += red[tid + o]; __syncthreads(); }
    float inv = 1.f / red[0];
    float* wp = weights + (size_t)b * TT;
    wp[tid] = e0 * inv; wp[tid + 256] = e1 * inv; wp[tid + 512] = e2 * inv; wp[tid + 768] = e3 * inv;
}

// ---------------------------------------------------------------------------
// 5) context[b][e] = sum_t weights[b][t]*encoded[b][t][e]
//    weights row staged into LDS via TDM tensor_load_to_lds (1D descriptor),
//    encoded streamed (second pass: L2-resident, 128MB < 192MB L2).
// ---------------------------------------------------------------------------
__global__ __launch_bounds__(128) void wa_context(const float* __restrict__ encoded,
                                                  const float* __restrict__ weights,
                                                  float* __restrict__ ctx) {
    __shared__ alignas(16) float wsh[TT];
    const int b = blockIdx.x >> 2, ec = blockIdx.x & 3, tid = threadIdx.x;

    if (tid < 32) {  // one wave issues the TDM copy: 1024 f32, 1D tile
        uint64_t ga  = (uint64_t)(uintptr_t)(weights + (size_t)b * TT);
        uint32_t lds = (uint32_t)(uintptr_t)&wsh[0];      // low 32 bits = LDS offset
        u32x4 g0;
        g0.x = 1u;                                        // count=1, user descriptor
        g0.y = lds;                                       // lds_addr
        g0.z = (uint32_t)ga;                              // global_addr[31:0]
        g0.w = (uint32_t)((ga >> 32) & 0x1FFFFFFu) | (2u << 30); // addr[56:32] | type=2
        u32x8 g1;
        g1.s0 = 2u << 16;                                 // data_size=4B
        g1.s1 = (uint32_t)TT << 16;                       // tensor_dim0[15:0]=1024
        g1.s2 = 1u << 16;                                 // tensor_dim0 hi=0, tensor_dim1=1
        g1.s3 = (uint32_t)TT << 16;                       // tensor_dim1 hi=0, tile_dim0=1024
        g1.s4 = 0u;                                       // tile_dim1/2 unused
        g1.s5 = (uint32_t)TT;                             // tensor_dim0_stride=1024
        g1.s6 = 0u;  g1.s7 = 0u;
        asm volatile("tensor_load_to_lds %0, %1" :: "s"(g0), "s"(g1) : "memory");
        __builtin_amdgcn_s_wait_tensorcnt(0);
    }
    __syncthreads();

    const int e = ec * 128 + tid;
    const float* ep = encoded + (size_t)b * TT * EE + e;
    float acc = 0.f;
    #pragma unroll 4
    for (int t = 0; t < TT; ++t)
        acc = fmaf(wsh[t], ep[(size_t)t * EE], acc);
    ctx[b * EE + e] = acc;
}

// ---------------------------------------------------------------------------
// 6) GRU gate GEMMs (bf16 WMMA): gi = x @ W_ih^T, gh = states @ W_hh^T
//    x = [inputs | context]. W rows are contiguous over K => direct B-fragments.
//    grid = (12 n-groups, 4 m-tiles); 8 waves/block, one (m,n) tile per wave.
// ---------------------------------------------------------------------------
__global__ __launch_bounds__(256) void wa_gates(const float* __restrict__ inputs,
                                                const float* __restrict__ context,
                                                const float* __restrict__ states,
                                                const float* __restrict__ W_ih,
                                                const float* __restrict__ W_hh,
                                                float* __restrict__ gi,
                                                float* __restrict__ gh) {
    __shared__ alignas(32) unsigned short xfrag[32 * 512];  // 16 rows x K=1024 (32KB)
    __shared__ alignas(32) unsigned short sfrag[16 * 512];  // 16 rows x K=512  (16KB)

    const int tid = threadIdx.x;
    const int mt  = blockIdx.y;                             // 16-row batch tile

    // stage x = [inputs|context] rows (bf16, A-fragment order)
    #pragma unroll 2
    for (int it = 0; it < 32; ++it) {
        int pidx = it * 256 + tid;                          // 8192 pairs
        int m = pidx >> 9, k = (pidx & 511) * 2;
        int brow = mt * 16 + m;
        const float2* src = (k < DIN)
            ? (const float2*)(inputs  + (size_t)brow * DIN + k)
            : (const float2*)(context + (size_t)brow * EE  + (k - DIN));
        float2 f = *src;
        *(uint32_t*)&xfrag[afrag_index(m, k)] =
            (uint32_t)f2bf(f.x) | ((uint32_t)f2bf(f.y) << 16);
    }
    // stage states rows
    #pragma unroll 2
    for (int it = 0; it < 16; ++it) {
        int pidx = it * 256 + tid;                          // 4096 pairs
        int m = pidx >> 8, k = (pidx & 255) * 2;
        float2 f = *(const float2*)(states + (size_t)(mt * 16 + m) * HH + k);
        *(uint32_t*)&sfrag[afrag_index(m, k)] =
            (uint32_t)f2bf(f.x) | ((uint32_t)f2bf(f.y) << 16);
    }
    __syncthreads();

    const int w = tid >> 5, lane = tid & 31, hf = lane >> 4, nl = lane & 15;
    const int j = (blockIdx.x * 8 + w) * 16 + nl;           // output column (0..1535)

    v8f ci = {};
    #pragma unroll 2
    for (int s = 0; s < 32; ++s) {
        const bf16x16 a = *(const bf16x16*)&xfrag[s * 512 + lane * 16];
        BF16Frag bw;
        const float* wr = W_ih + (size_t)j * KX + s * 32 + hf * 16;
        #pragma unroll
        for (int i = 0; i < 16; ++i) bw.s[i] = f2bf(wr[i]);
        ci = __builtin_amdgcn_wmma_f32_16x16x32_bf16(false, a, false, bw.v,
                                                     (short)0, ci, false, false);
    }
    v8f ch = {};
    #pragma unroll 2
    for (int s = 0; s < 16; ++s) {
        const bf16x16 a = *(const bf16x16*)&sfrag[s * 512 + lane * 16];
        BF16Frag bw;
        const float* wr = W_hh + (size_t)j * HH + s * 32 + hf * 16;
        #pragma unroll
        for (int i = 0; i < 16; ++i) bw.s[i] = f2bf(wr[i]);
        ch = __builtin_amdgcn_wmma_f32_16x16x32_bf16(false, a, false, bw.v,
                                                     (short)0, ch, false, false);
    }

    #pragma unroll
    for (int r = 0; r < 8; ++r) {
        int row = mt * 16 + r + hf * 8;
        gi[(size_t)row * G3 + j] = ci[r];
        gh[(size_t)row * G3 + j] = ch[r];
    }
}

// ---------------------------------------------------------------------------
// 7) elementwise GRU gates -> output h'
// ---------------------------------------------------------------------------
__global__ __launch_bounds__(256) void wa_gru_out(const float* __restrict__ gi,
                                                  const float* __restrict__ gh,
                                                  const float* __restrict__ b_ih,
                                                  const float* __restrict__ b_hh,
                                                  const float* __restrict__ states,
                                                  float* __restrict__ out) {
    int g = blockIdx.x * 256 + threadIdx.x;                 // 32768
    int b = g >> 9, h = g & 511;
    const float* gib = gi + (size_t)b * G3;
    const float* ghb = gh + (size_t)b * G3;
    float i_r = gib[h]        + b_ih[h];
    float i_z = gib[h + 512]  + b_ih[h + 512];
    float i_n = gib[h + 1024] + b_ih[h + 1024];
    float h_r = ghb[h]        + b_hh[h];
    float h_z = ghb[h + 512]  + b_hh[h + 512];
    float h_n = ghb[h + 1024] + b_hh[h + 1024];
    float r = 1.f / (1.f + __expf(-(i_r + h_r)));
    float z = 1.f / (1.f + __expf(-(i_z + h_z)));
    float n = FAST_TANH(i_n + r * h_n);
    out[g] = (1.f - z) * n + z * states[g];
}

// ---------------------------------------------------------------------------
// launcher
// ---------------------------------------------------------------------------
extern "C" void kernel_launch(void* const* d_in, const int* in_sizes, int n_in,
                              void* d_out, int out_size, void* d_ws, size_t ws_size,
                              hipStream_t stream) {
    (void)in_sizes; (void)n_in; (void)out_size; (void)ws_size;
    const float* inputs  = (const float*)d_in[0];
    const float* states  = (const float*)d_in[1];
    const float* encoded = (const float*)d_in[2];
    const float* W_enc   = (const float*)d_in[3];
    const float* W_state = (const float*)d_in[4];
    const float* vvec    = (const float*)d_in[5];
    const float* W_ih    = (const float*)d_in[6];
    const float* W_hh    = (const float*)d_in[7];
    const float* b_ih    = (const float*)d_in[8];
    const float* b_hh    = (const float*)d_in[9];
    float* out = (float*)d_out;

    char* ws = (char*)d_ws;                       // 2 MB total
    unsigned short* Wt = (unsigned short*)(ws);               //  512 KB  bf16 W_enc^T
    float* stp     = (float*)(ws + 524288);                   //  128 KB
    float* scores  = (float*)(ws + 655360);                   //  256 KB
    float* weights = (float*)(ws + 917504);                   //  256 KB
    float* ctx     = (float*)(ws + 1179648);                  //  128 KB
    float* gi      = (float*)(ws + 1310720);                  //  384 KB
    float* gh      = (float*)(ws + 1703936);                  //  384 KB

    wa_prep_wt <<<1024, 256, 0, stream>>>(W_enc, Wt);
    wa_stproj  <<<128,  256, 0, stream>>>(states, W_state, stp);
    wa_scores  <<<1024, 128, 0, stream>>>(encoded, Wt, stp, vvec, scores);
    wa_softmax <<<BB,   256, 0, stream>>>(scores, weights);
    wa_context <<<256,  128, 0, stream>>>(encoded, weights, ctx);
    wa_gates   <<<dim3(12, 4), 256, 0, stream>>>(inputs, ctx, states, W_ih, W_hh, gi, gh);
    wa_gru_out <<<128,  256, 0, stream>>>(gi, gh, b_ih, b_hh, states, out);
}
